// JQSANIModel_21878563406027
// MI455X (gfx1250) — matrix operand order repaired and checked by reading, not compiled, not run on hardware
//
#include <hip/hip_runtime.h>
#include <hip/hip_bf16.h>
#include <cstdint>

typedef __attribute__((ext_vector_type(16))) _Float16 v16h;
typedef __attribute__((ext_vector_type(8)))  _Float16 v8h;
typedef __attribute__((ext_vector_type(8)))  float    v8f;

#define N_MOL 512
#define N_ATM 64
#define A_TOT (N_MOL * N_ATM)
#define L_AEV 384
#define N_EXP 8
#define H1 160
#define H2 128
#define H3 96
#define KP1 416   // 386 padded to multiple of 32
#define MTILE 32  // rows (atoms) per wave: two 16x16 M-tiles sharing B fragments
#define A0_CONST 0.529177249f

__constant__ float c_SIGMA[8] = {0.5515909f, 1.8886297f, 1.3225029f, 1.2316629f,
                                 2.1884933f, 1.7750372f, 1.3677907f, 1.3820058f};

// ---------------- math helpers ----------------
// branchless CELU(alpha=0.1): max(x,0) + min(0.1*(exp(10x)-1), 0)
// large x -> __expf = +inf -> min(inf,0) = 0; very negative x -> -0.1. No branches.
__device__ __forceinline__ float celu01(float x) {
    float e = __expf(10.f * x);
    return fmaxf(x, 0.f) + fminf(0.1f * (e - 1.f), 0.f);
}
__device__ __forceinline__ float softplusf(float x) {
    return x > 20.f ? x : log1pf(expf(x));
}

// ---------------- WMMA fragment loaders ----------------
// A fragment (16x32 f16, M x K): lane l<16 holds row l, K = k0+{0..7} and k0+{16..23};
// lane l+16 holds row l, K = k0+{8..15} and k0+{24..31}.  (ISA 7.12.2)
__device__ __forceinline__ v16h load_a_frag(const _Float16* tile, int pitch, int k0, int lane) {
    int l  = lane & 15;
    int hi = (lane >> 4) & 1;
    const _Float16* p = tile + l * pitch + k0 + hi * 8;
    v8h lo  = *(const v8h*)(p);
    v8h hi8 = *(const v8h*)(p + 16);
    return __builtin_shufflevector(lo, hi8, 0,1,2,3,4,5,6,7,8,9,10,11,12,13,14,15);
}

// B fragment (32x16 f16, K x N) from f16 W^T stored [n][Kpad]:
// lane c<16 holds column n0+c, K = k0+{0..7} and k0+{16..23}; lane c+16 holds K = k0+{8..15} and k0+{24..31}.
__device__ __forceinline__ v16h load_b_frag(const _Float16* WT, int Kpad, int n0, int k0, int lane) {
    int c  = lane & 15;
    int hi = (lane >> 4) & 1;
    const _Float16* p = WT + (size_t)(n0 + c) * Kpad + k0 + hi * 8;
    v8h b0 = *(const v8h*)(p);
    v8h b1 = *(const v8h*)(p + 16);
    return __builtin_shufflevector(b0, b1, 0,1,2,3,4,5,6,7,8,9,10,11,12,13,14,15);
}

// One dense layer for a 32-row tile: out = CELU(in[32xK] @ W[KxN] + b)
// inT pitch == K, outT pitch == N.  Two 16-row M-tiles share every B fragment.
// Full wave (EXEC all ones) required for the WMMAs.
__device__ void mlp_layer(const _Float16* __restrict__ WT, const float* __restrict__ bias,
                          const _Float16* __restrict__ inT, _Float16* __restrict__ outT,
                          int K, int N, int lane) {
    int c  = lane & 15;
    int hi = (lane >> 4) & 1;
    const _Float16* inT1 = inT + 16 * K;   // rows 16..31
#pragma unroll 1
    for (int n0 = 0; n0 < N; n0 += 16) {
        v8f acc0 = {};
        v8f acc1 = {};
#pragma unroll 4
        for (int k0 = 0; k0 < K; k0 += 32) {
            v16h b  = load_b_frag(WT, K, n0, k0, lane);
            v16h a0 = load_a_frag(inT,  K, k0, lane);
            v16h a1 = load_a_frag(inT1, K, k0, lane);
            acc0 = __builtin_amdgcn_wmma_f32_16x16x32_f16(false, a0, false, b,
                                                          (short)0, acc0, false, false);
            acc1 = __builtin_amdgcn_wmma_f32_16x16x32_f16(false, a1, false, b,
                                                          (short)0, acc1, false, false);
        }
        float bv = bias[n0 + c];
#pragma unroll
        for (int j = 0; j < 8; j++) {
            // C frag: M = hi*8+j, N = n0+c
            outT[(hi * 8 + j) * N + n0 + c]        = (_Float16)celu01(acc0[j] + bv);
            outT[(16 + hi * 8 + j) * N + n0 + c]   = (_Float16)celu01(acc1[j] + bv);
        }
    }
}

// vectorized dot over a 96-wide f16 LDS row against an f16 weight vector
__device__ __forceinline__ float dot96(const _Float16* row, const _Float16* wv) {
    float acc = 0.f;
#pragma unroll
    for (int k0 = 0; k0 < H3; k0 += 8) {
        v8h h = *(const v8h*)(row + k0);
        v8h g = *(const v8h*)(wv + k0);
#pragma unroll
        for (int i = 0; i < 8; i++) acc += (float)h[i] * (float)g[i];
    }
    return acc;
}

// ---------------- weight prep: f32 [e][K][N] -> f16 [e][n][Kpad] (zero-padded K) --------
__global__ void prep_wt(const float* __restrict__ src, _Float16* __restrict__ dst,
                        int E, int K, int N, int Kpad) {
    int total = E * N * Kpad;
    for (int idx = blockIdx.x * blockDim.x + threadIdx.x; idx < total;
         idx += gridDim.x * blockDim.x) {
        int e = idx / (N * Kpad);
        int r = idx % (N * Kpad);
        int n = r / Kpad;
        int k = r % Kpad;
        float v = (k < K) ? src[(size_t)e * K * N + (size_t)k * N + n] : 0.f;
        dst[idx] = (_Float16)v;
    }
}

// f32 -> f16 copy (for W4 vectors)
__global__ void cvt_f16(const float* __restrict__ src, _Float16* __restrict__ dst, int n) {
    int i = blockIdx.x * blockDim.x + threadIdx.x;
    if (i < n) dst[i] = (_Float16)src[i];
}

// ---------------- chi network: A x (384->160->128->96->1), softplus ----------------
__global__ void chi_mlp(const float* __restrict__ aev, const int* __restrict__ species,
                        const _Float16* __restrict__ W1T, const _Float16* __restrict__ W2T,
                        const _Float16* __restrict__ W3T, const _Float16* __restrict__ W4h,
                        const float* __restrict__ b1, const float* __restrict__ b2,
                        const float* __restrict__ b3, const float* __restrict__ b4s,
                        float* __restrict__ chi) {
    __shared__ __align__(16) _Float16 tA[MTILE * L_AEV];
    __shared__ __align__(16) _Float16 t1[MTILE * H1];
    __shared__ __align__(16) _Float16 t2[MTILE * H2];
    __shared__ __align__(16) _Float16 t3[MTILE * H3];
    int lane = threadIdx.x;
    int base = blockIdx.x * MTILE;

    if (lane == 0 && base + MTILE < A_TOT)
        __builtin_prefetch(&aev[(size_t)(base + MTILE) * L_AEV], 0, 1);

    for (int m = 0; m < MTILE; m++) {
        int atom = base + m;
        for (int k = lane; k < L_AEV; k += 32)
            tA[m * L_AEV + k] = (_Float16)aev[(size_t)atom * L_AEV + k];
    }
    __syncthreads();
    mlp_layer(W1T, b1, tA, t1, L_AEV, H1, lane);
    __syncthreads();
    mlp_layer(W2T, b2, t1, t2, H1, H2, lane);
    __syncthreads();
    mlp_layer(W3T, b3, t2, t3, H2, H3, lane);
    __syncthreads();
    {
        int atom = base + lane;   // 32 lanes, 32 atoms
        float v = softplusf(dot96(t3 + lane * H3, W4h) + b4s[0]);
        chi[atom] = (species[atom] != -1) ? v : 0.f;
    }
}

// ---------------- per-molecule: q equilibration + erf-screened ESP ----------------
__device__ __forceinline__ float block_reduce64(float v, float* red, int i) {
    red[i] = v;
    __syncthreads();
#pragma unroll
    for (int s = 32; s > 0; s >>= 1) {
        if (i < s) red[i] += red[i + s];
        __syncthreads();
    }
    float r = red[0];
    __syncthreads();
    return r;
}

__global__ void molecule_kernel(const int* __restrict__ species, const float* __restrict__ coords,
                                const float* __restrict__ Qnet, const float* __restrict__ chi,
                                float* __restrict__ qws, float* __restrict__ espws,
                                float* __restrict__ coulws, float* __restrict__ qout) {
    int b = blockIdx.x;
    int i = threadIdx.x;  // 64 threads
    __shared__ float red[N_ATM];
    __shared__ float sx[N_ATM], sy[N_ATM], sz[N_ATM], sq[N_ATM], ssig[N_ATM];

    int  gidx = b * N_ATM + i;
    int  sp   = species[gidx];
    float padi = (sp != -1) ? 1.f : 0.f;
    float chi_i = chi[gidx];

    float chi_sum = block_reduce64(chi_i, red, i);
    float num     = block_reduce64(padi,  red, i);

    float Q = Qnet[b];
    float k_net = 1.f + fabsf(Q) / chi_sum;
    float chi_mean = chi_sum / num;
    float k_p = (Q > 0.f) ? k_net : 1.f;
    float k_n = (Q < 0.f) ? k_net : 1.f;
    float qi = (-k_n * chi_i + k_p * chi_mean) * padi;

    sq[i] = qi;
    sx[i] = coords[(size_t)gidx * 3 + 0];
    sy[i] = coords[(size_t)gidx * 3 + 1];
    sz[i] = coords[(size_t)gidx * 3 + 2];
    int spc = sp < 0 ? 0 : sp;
    ssig[i] = c_SIGMA[spc];
    __syncthreads();

    float sig2_i = ssig[i] * ssig[i];
    float xi = sx[i], yi = sy[i], zi = sz[i];
    float esp = 0.f;
    for (int j = 0; j < N_ATM; j++) {
        if (j == i) continue;
        float dx = xi - sx[j], dy = yi - sy[j], dz = zi - sz[j];
        float d  = sqrtf(dx * dx + dy * dy + dz * dz + 1e-16f) / A0_CONST;
        float ss = fmaxf(sig2_i + ssig[j] * ssig[j], 1e-8f);
        esp += sq[j] * erff(d * rsqrtf(2.f * ss)) / d;
    }
    esp *= padi;

    qws[gidx]   = qi;
    espws[gidx] = esp;
    qout[gidx]  = qi;

    float cpart = block_reduce64(qi * esp, red, i);
    if (i == 0) coulws[b] = 0.5f * cpart;
}

// ---------------- deterministic species bucketing (ballot compaction) ----------------
__global__ void bucket_kernel(const int* __restrict__ species, int* __restrict__ cnt,
                              int* __restrict__ idxlist) {
    int e = blockIdx.x;      // 8 blocks
    int lane = threadIdx.x;  // 32 lanes
    int count = 0;
    for (int base = 0; base < A_TOT; base += 32) {
        int atom = base + lane;
        bool m = (species[atom] == e);
        unsigned b32 = (unsigned)__ballot(m);
        if (m) {
            int off = __popc(b32 & ((1u << lane) - 1u));
            idxlist[e * A_TOT + count + off] = atom;
        }
        count += __popc(b32);
    }
    if (lane == 0) cnt[e] = count;
}

// ---------------- expert MLPs: routed [aev | q | esp] (386 -> pad 416) ----------------
__global__ void expert_mlp(const float* __restrict__ aev, const float* __restrict__ qws,
                           const float* __restrict__ espws,
                           const _Float16* __restrict__ W1T, const _Float16* __restrict__ W2T,
                           const _Float16* __restrict__ W3T,
                           const float* __restrict__ b1, const float* __restrict__ b2,
                           const float* __restrict__ b3, const _Float16* __restrict__ W4h,
                           const float* __restrict__ b4,
                           const int* __restrict__ cnt, const int* __restrict__ idxlist,
                           float* __restrict__ energy) {
    __shared__ __align__(16) _Float16 tA[MTILE * KP1];
    __shared__ __align__(16) _Float16 t1[MTILE * H1];
    __shared__ __align__(16) _Float16 t2[MTILE * H2];
    __shared__ __align__(16) _Float16 t3[MTILE * H3];
    __shared__ int sIdx[MTILE];
    int lane = threadIdx.x;

    int tiles[N_EXP], cn[N_EXP], T = 0;
#pragma unroll
    for (int e = 0; e < N_EXP; e++) {
        cn[e] = cnt[e];
        tiles[e] = (cn[e] + MTILE - 1) / MTILE;
        T += tiles[e];
    }

    for (int t = blockIdx.x; t < T; t += gridDim.x) {
        int e = 0, rem = t;
        while (rem >= tiles[e]) { rem -= tiles[e]; e++; }
        int base = rem * MTILE;
        int ce = cn[e];

        sIdx[lane] = (base + lane < ce) ? idxlist[e * A_TOT + base + lane] : -1;
        __syncthreads();

        for (int m = 0; m < MTILE; m++) {
            int atom = sIdx[m];
            if (atom >= 0) {
                for (int k = lane; k < L_AEV; k += 32)
                    tA[m * KP1 + k] = (_Float16)aev[(size_t)atom * L_AEV + k];
                if (lane == 0) {
                    tA[m * KP1 + 384] = (_Float16)qws[atom];
                    tA[m * KP1 + 385] = (_Float16)espws[atom];
                }
                if (lane < KP1 - 386) tA[m * KP1 + 386 + lane] = (_Float16)0.f;
            } else {
                for (int k = lane; k < KP1; k += 32) tA[m * KP1 + k] = (_Float16)0.f;
            }
        }
        __syncthreads();
        mlp_layer(W1T + (size_t)e * H1 * KP1, b1 + e * H1, tA, t1, KP1, H1, lane);
        __syncthreads();
        mlp_layer(W2T + (size_t)e * H2 * H1, b2 + e * H2, t1, t2, H1, H2, lane);
        __syncthreads();
        mlp_layer(W3T + (size_t)e * H3 * H2, b3 + e * H3, t2, t3, H2, H3, lane);
        __syncthreads();
        {
            int atom = sIdx[lane];   // 32 lanes, 32 atoms
            if (atom >= 0)
                energy[atom] = dot96(t3 + lane * H3, W4h + e * H3) + b4[e];
        }
        __syncthreads();
    }
}

// ---------------- finalize: species passthrough + molecular energy ----------------
__global__ void finalize_kernel(const int* __restrict__ species, const float* __restrict__ energy,
                                const float* __restrict__ coulws, int* __restrict__ outSp,
                                float* __restrict__ outE) {
    int b = blockIdx.x;
    int i = threadIdx.x;
    __shared__ float red[N_ATM];
    int gidx = b * N_ATM + i;
    outSp[gidx] = species[gidx];
    float s = block_reduce64(energy[gidx], red, i);
    if (i == 0) outE[b] = s + coulws[b];
}

// ---------------- host orchestration ----------------
extern "C" void kernel_launch(void* const* d_in, const int* in_sizes, int n_in,
                              void* d_out, int out_size, void* d_ws, size_t ws_size,
                              hipStream_t stream) {
    const int*   species = (const int*)d_in[0];
    const float* coords  = (const float*)d_in[1];
    const float* Qnet    = (const float*)d_in[2];
    const float* aev     = (const float*)d_in[3];
    const float* chi_W1  = (const float*)d_in[4];
    const float* chi_b1  = (const float*)d_in[5];
    const float* chi_W2  = (const float*)d_in[6];
    const float* chi_b2  = (const float*)d_in[7];
    const float* chi_W3  = (const float*)d_in[8];
    const float* chi_b3  = (const float*)d_in[9];
    const float* chi_W4  = (const float*)d_in[10];
    const float* chi_b4  = (const float*)d_in[11];
    const float* W1      = (const float*)d_in[12];
    const float* b1      = (const float*)d_in[13];
    const float* W2      = (const float*)d_in[14];
    const float* b2      = (const float*)d_in[15];
    const float* W3      = (const float*)d_in[16];
    const float* b3      = (const float*)d_in[17];
    const float* W4      = (const float*)d_in[18];
    const float* b4      = (const float*)d_in[19];

    char* w = (char*)d_ws;
    size_t off = 0;
    auto carve = [&](size_t bytes) -> char* {
        char* p = w + off;
        off = (off + bytes + 255) & ~(size_t)255;
        return p;
    };
    _Float16* chiW1T = (_Float16*)carve((size_t)H1 * L_AEV * 2);
    _Float16* chiW2T = (_Float16*)carve((size_t)H2 * H1 * 2);
    _Float16* chiW3T = (_Float16*)carve((size_t)H3 * H2 * 2);
    _Float16* chiW4h = (_Float16*)carve((size_t)H3 * 2);
    _Float16* W1T    = (_Float16*)carve((size_t)N_EXP * H1 * KP1 * 2);
    _Float16* W2T    = (_Float16*)carve((size_t)N_EXP * H2 * H1 * 2);
    _Float16* W3T    = (_Float16*)carve((size_t)N_EXP * H3 * H2 * 2);
    _Float16* W4h    = (_Float16*)carve((size_t)N_EXP * H3 * 2);
    float* chiA   = (float*)carve((size_t)A_TOT * 4);
    float* qws    = (float*)carve((size_t)A_TOT * 4);
    float* espws  = (float*)carve((size_t)A_TOT * 4);
    float* coulws = (float*)carve((size_t)N_MOL * 4);
    float* energy = (float*)carve((size_t)A_TOT * 4);
    int*   cnt    = (int*)carve((size_t)N_EXP * 4);
    int*   idxl   = (int*)carve((size_t)N_EXP * A_TOT * 4);
    (void)ws_size; (void)in_sizes; (void)n_in; (void)out_size;

    int*   outSp = (int*)d_out;
    float* outE  = (float*)d_out + A_TOT;
    float* outQ  = (float*)d_out + A_TOT + N_MOL;

    // 1) weight conversion (f32 [e][K][N] -> f16 [e][n][Kpad])
    prep_wt<<<256, 256, 0, stream>>>(chi_W1, chiW1T, 1, L_AEV, H1, L_AEV);
    prep_wt<<<96, 256, 0, stream>>>(chi_W2, chiW2T, 1, H1, H2, H1);
    prep_wt<<<48, 256, 0, stream>>>(chi_W3, chiW3T, 1, H2, H3, H2);
    prep_wt<<<1, 128, 0, stream>>>(chi_W4, chiW4h, 1, H3, 1, H3);
    prep_wt<<<1024, 256, 0, stream>>>(W1, W1T, N_EXP, 386, H1, KP1);
    prep_wt<<<640, 256, 0, stream>>>(W2, W2T, N_EXP, H1, H2, H1);
    prep_wt<<<384, 256, 0, stream>>>(W3, W3T, N_EXP, H2, H3, H2);
    cvt_f16<<<3, 256, 0, stream>>>(W4, W4h, N_EXP * H3);

    // 2) chi net (WMMA)
    chi_mlp<<<A_TOT / MTILE, 32, 0, stream>>>(aev, species, chiW1T, chiW2T, chiW3T, chiW4h,
                                              chi_b1, chi_b2, chi_b3, chi_b4, chiA);

    // 3) per-molecule charge equilibration + ESP
    molecule_kernel<<<N_MOL, N_ATM, 0, stream>>>(species, coords, Qnet, chiA,
                                                 qws, espws, coulws, outQ);

    // 4) deterministic species bucketing + energy init
    bucket_kernel<<<N_EXP, 32, 0, stream>>>(species, cnt, idxl);
    hipMemsetAsync(energy, 0, (size_t)A_TOT * 4, stream);

    // 5) routed expert MLPs (WMMA)
    expert_mlp<<<1024, 32, 0, stream>>>(aev, qws, espws, W1T, W2T, W3T,
                                        b1, b2, b3, W4h, b4, cnt, idxl, energy);

    // 6) finalize
    finalize_kernel<<<N_MOL, N_ATM, 0, stream>>>(species, energy, coulws, outSp, outE);
}
